// DeathModel_46514495815922
// MI455X (gfx1250) — matrix operand order repaired and checked by reading, not compiled, not run on hardware
//
#include <hip/hip_runtime.h>

typedef __attribute__((ext_vector_type(16))) __bf16 v16bf;
typedef __attribute__((ext_vector_type(8)))  float  v8f;

#define B_  128
#define T_  256
#define D_  512
#define H_  1024
#define G_  4096   // 4H
#define KC  1536   // D + H (concatenated K)
#define KT  48     // KC / 32 (total K-tiles)
#define KTH 24     // K-tiles per split-K half
#define NT  256    // G_ / 16

// Async global->LDS copy via inline asm (CDNA5 GLOBAL_LOAD_ASYNC_TO_LDS_B128,
// tracked by ASYNCcnt).  Set to 0 to fall back to load + ds_store.
#define ASYNC_USE 1

__device__ __forceinline__ unsigned short f2bf(float f) {
  unsigned int u = __float_as_uint(f);
  u += 0x7FFFu + ((u >> 16) & 1u);          // round-to-nearest-even
  return (unsigned short)(u >> 16);
}

__device__ __forceinline__ float sigf(float x) { return 1.0f / (1.0f + __expf(-x)); }

// ---------------------------------------------------------------------------
// Pack [W ; U] (fp32, row-major KxN, K=1536, N=4096) into bf16 WMMA B-fragment
// order.  Tile (kt, nt) covers K=[kt*32, +32), N=[nt*16, +16).  Contiguous-k
// 32x16 B layout: lane n(0..15) holds column n, K=k0+e; lane n+16 holds column
// n, K=k0+16+e.  Lane l's 16 bf16 stored contiguously at tile*1024 + l*32 B.
// ---------------------------------------------------------------------------
__global__ __launch_bounds__(256) void pack_weights(
    const float* __restrict__ Wk, const float* __restrict__ Wr,
    unsigned short* __restrict__ Bp) {
  int idx = blockIdx.x * blockDim.x + threadIdx.x;
  if (idx >= KT * NT * 32 * 16) return;
  int e    = idx & 15;
  int lane = (idx >> 4) & 31;
  int tile = idx >> 9;
  int kt   = tile / NT;
  int nt   = tile - kt * NT;
  int kl   = (lane < 16) ? e : (16 + e);
  int Kg   = kt * 32 + kl;
  int col  = nt * 16 + (lane & 15);
  float v  = (Kg < D_) ? Wk[(size_t)Kg * G_ + col]
                       : Wr[(size_t)(Kg - D_) * G_ + col];
  Bp[(size_t)tile * 512 + lane * 16 + e] = f2bf(v);
}

__global__ __launch_bounds__(256) void convert_x(
    const float* __restrict__ x, unsigned short* __restrict__ xbf, int n) {
  int idx = blockIdx.x * blockDim.x + threadIdx.x;
  if (idx < n) xbf[idx] = f2bf(x[idx]);
}

__global__ __launch_bounds__(256) void init_state(
    const unsigned short* __restrict__ xbf, unsigned short* __restrict__ Abuf,
    float* __restrict__ c, float* __restrict__ hf) {
  int idx = blockIdx.x * blockDim.x + threadIdx.x;   // over B_*KC
  if (idx >= B_ * KC) return;
  int b   = idx / KC;
  int col = idx - b * KC;
  Abuf[idx] = (col < D_) ? xbf[((size_t)b * T_) * D_ + col] : (unsigned short)0;
  if (idx < B_ * H_) { c[idx] = 0.0f; hf[idx] = 0.0f; }
}

// ---------------------------------------------------------------------------
// Fragment helpers
// ---------------------------------------------------------------------------
__device__ __forceinline__ v16bf load_a_frag(
    const unsigned short* __restrict__ arow, int kt, int halfSel) {
  union { uint4 q[2]; v16bf v; } ua;
  const unsigned short* p = arow + kt * 32 + (halfSel ? 16 : 0);
  ua.q[0] = *(const uint4*)(p);
  ua.q[1] = *(const uint4*)(p + 8);
  return ua.v;
}

// Copy one 16-byte quantum of the current B chunk into LDS.
__device__ __forceinline__ void stage16(unsigned short* __restrict__ dst,
                                        const unsigned short* __restrict__ src) {
#if ASYNC_USE
  // VDST = per-lane LDS byte address (low 32 bits of the generic pointer),
  // VADDR = 64-bit global VA.  Tracked by ASYNCcnt.
  unsigned int laddr = (unsigned int)(uintptr_t)dst;
  asm volatile("global_load_async_to_lds_b128 %0, %1, off"
               :: "v"(laddr), "v"(src) : "memory");
#else
  *(uint4*)dst = *(const uint4*)src;
#endif
}

__device__ __forceinline__ void wait_async_staged() {
#if ASYNC_USE
  asm volatile("s_wait_asynccnt 0x0" ::: "memory");
#endif
}

// 4 WMMAs consuming B fragments from an LDS buffer (2048 ushorts = 4 tiles).
// All 8 ds_load_b128 issued before the WMMAs so waits are partial/overlapped.
__device__ __forceinline__ void mma4_lds(v8f* __restrict__ acc, v16bf a,
                                         const unsigned short* __restrict__ buf,
                                         int lane) {
  uint4 q[8];
#pragma unroll
  for (int j = 0; j < 4; ++j) {
    const unsigned short* p = buf + j * 512 + lane * 16;
    q[2 * j]     = *(const uint4*)(p);
    q[2 * j + 1] = *(const uint4*)(p + 8);
  }
#pragma unroll
  for (int j = 0; j < 4; ++j) {
    union { uint4 q[2]; v16bf v; } ub;
    ub.q[0] = q[2 * j];
    ub.q[1] = q[2 * j + 1];
    acc[j] = __builtin_amdgcn_wmma_f32_16x16x32_bf16(
        false, a, false, ub.v, (short)0, acc[j], false, false);
  }
}

// ---------------------------------------------------------------------------
// One LSTM step GEMM, split-K by 2:
//   zPart[kHalf] (128x4096 f32) = Abuf[:, kHalf*768 : +768] @ Bp[kHalf half]
// Grid (64, 2): blockIdx.x = 64-col strip, blockIdx.y = K half.
// 8 waves per block (one per 16-row tile).  B chunk (4 KB / K-step) staged
// cooperatively into LDS (double-buffered) -> each weight byte fetched from
// L2 exactly once per step.
// ---------------------------------------------------------------------------
__global__ __launch_bounds__(256) void lstm_step_gemm(
    const unsigned short* __restrict__ Abuf,
    const unsigned short* __restrict__ Bp,
    float* __restrict__ z0, float* __restrict__ z1) {
  __shared__ unsigned short sB[2][2048];   // 2 x 4 KB

  const int tid     = threadIdx.x;
  const int wq      = tid >> 5;       // wave in block == mTile (0..7)
  const int lane    = tid & 31;
  const int nStrip  = blockIdx.x;     // 0..63
  const int kHalf   = blockIdx.y;     // 0..1
  const int m0      = wq * 16;
  const int halfSel = lane >> 4;
  const int lrow    = lane & 15;

  v8f acc[4] = {};

  const int row = m0 + lrow;
  const unsigned short* arow  = Abuf + (size_t)row * KC + kHalf * (KTH * 32);
  const unsigned short* chunk =
      Bp + ((size_t)(kHalf * KTH) * NT + (size_t)nStrip * 4) * 512 + tid * 8;
  const size_t kstride = (size_t)NT * 512;   // ushorts between K-steps
  float* z = kHalf ? z1 : z0;

  // prologue
  stage16(&sB[0][tid * 8], chunk);
  v16bf a0 = load_a_frag(arow, 0, halfSel);

  for (int kt = 0; kt < KTH; kt += 2) {
    // ---- even sub-iteration: consume sB[0] ----
    wait_async_staged();
    __syncthreads();                                   // sB[0] ready
    stage16(&sB[1][tid * 8], chunk + (size_t)(kt + 1) * kstride);
    v16bf a1 = load_a_frag(arow, kt + 1, halfSel);
    __builtin_prefetch(chunk + (size_t)(kt + 2) * kstride, 0, 1);
    mma4_lds(acc, a0, sB[0], lane);
    __syncthreads();                                   // done reading sB[0]

    // ---- odd sub-iteration: consume sB[1] ----
    wait_async_staged();
    __syncthreads();                                   // sB[1] ready
    if (kt + 2 < KTH) {
      stage16(&sB[0][tid * 8], chunk + (size_t)(kt + 2) * kstride);
      a0 = load_a_frag(arow, kt + 2, halfSel);
    }
    mma4_lds(acc, a1, sB[1], lane);
    __syncthreads();                                   // done reading sB[1]
  }

  // store C: VGPR r, lanes 0-15 -> (M=r, N=lane); lanes 16-31 -> (M=r+8)
  const int n0 = nStrip * 64 + lrow;
#pragma unroll
  for (int r = 0; r < 8; ++r) {
    const int rr = m0 + r + halfSel * 8;
    float* zr = z + (size_t)rr * G_ + n0;
    zr[0]  = acc[0][r];
    zr[16] = acc[1][r];
    zr[32] = acc[2][r];
    zr[48] = acc[3][r];
  }
}

// ---------------------------------------------------------------------------
// Gate nonlinearities + state update (Keras gate order i,f,g,o along columns).
// Sums the two split-K partials, writes h (bf16) into Abuf cols 512..1535 and
// stages x_{t+1} into cols 0..511.
// ---------------------------------------------------------------------------
__global__ __launch_bounds__(256) void lstm_elem(
    const float* __restrict__ z0, const float* __restrict__ z1,
    const float* __restrict__ bias, float* __restrict__ c,
    unsigned short* __restrict__ Abuf, float* __restrict__ hf,
    const unsigned short* __restrict__ xbf, int t_next) {
  int idx = blockIdx.x * blockDim.x + threadIdx.x;
  if (idx >= B_ * H_) return;
  int b = idx >> 10;
  int j = idx & (H_ - 1);
  const float* zr0 = z0 + (size_t)b * G_;
  const float* zr1 = z1 + (size_t)b * G_;
  float iv = sigf(zr0[j]           + zr1[j]           + bias[j]);
  float fv = sigf(zr0[j + H_]      + zr1[j + H_]      + bias[j + H_]);
  float gv = tanhf(zr0[j + 2 * H_] + zr1[j + 2 * H_]  + bias[j + 2 * H_]);
  float ov = sigf(zr0[j + 3 * H_]  + zr1[j + 3 * H_]  + bias[j + 3 * H_]);
  float cn = fv * c[idx] + iv * gv;
  c[idx]   = cn;
  float hn = ov * tanhf(cn);
  hf[idx]  = hn;
  Abuf[(size_t)b * KC + D_ + j] = f2bf(hn);

  if (t_next < T_ && idx < B_ * D_) {
    int b2  = idx >> 9;
    int col = idx & (D_ - 1);
    Abuf[(size_t)b2 * KC + col] = xbf[((size_t)b2 * T_ + t_next) * D_ + col];
  }
}

// ---------------------------------------------------------------------------
// out[b] = sigmoid(h_last[b] . dense_w + dense_b)
// ---------------------------------------------------------------------------
__global__ __launch_bounds__(256) void final_dense(
    const float* __restrict__ hf, const float* __restrict__ w,
    const float* __restrict__ bptr, float* __restrict__ out) {
  __shared__ float red[256];
  int b = blockIdx.x;
  float s = 0.0f;
  for (int j = threadIdx.x; j < H_; j += 256) s += hf[(size_t)b * H_ + j] * w[j];
  red[threadIdx.x] = s;
  __syncthreads();
  for (int st = 128; st > 0; st >>= 1) {
    if (threadIdx.x < st) red[threadIdx.x] += red[threadIdx.x + st];
    __syncthreads();
  }
  if (threadIdx.x == 0) out[b] = sigf(red[0] + bptr[0]);
}

extern "C" void kernel_launch(void* const* d_in, const int* in_sizes, int n_in,
                              void* d_out, int out_size, void* d_ws, size_t ws_size,
                              hipStream_t stream) {
  const float* x    = (const float*)d_in[0];   // [B, T, D]
  const float* Wk   = (const float*)d_in[1];   // [D, 4H]
  const float* Wr   = (const float*)d_in[2];   // [H, 4H]
  const float* bias = (const float*)d_in[3];   // [4H]
  const float* dw   = (const float*)d_in[4];   // [H, 1]
  const float* db   = (const float*)d_in[5];   // [1]
  float* out = (float*)d_out;                  // [B, 1] fp32

  char* ws = (char*)d_ws;
  size_t off = 0;
  unsigned short* Bp   = (unsigned short*)(ws + off); off += (size_t)KT * NT * 1024;
  unsigned short* xbf  = (unsigned short*)(ws + off); off += (size_t)B_ * T_ * D_ * 2;
  unsigned short* Abuf = (unsigned short*)(ws + off); off += (size_t)B_ * KC * 2;
  float* c  = (float*)(ws + off); off += (size_t)B_ * H_ * 4;
  float* hf = (float*)(ws + off); off += (size_t)B_ * H_ * 4;
  float* z0 = (float*)(ws + off); off += (size_t)B_ * G_ * 4;
  float* z1 = (float*)(ws + off); off += (size_t)B_ * G_ * 4;

  const int nX = B_ * T_ * D_;
  pack_weights<<<(KT * NT * 512 + 255) / 256, 256, 0, stream>>>(Wk, Wr, Bp);
  convert_x<<<(nX + 255) / 256, 256, 0, stream>>>(x, xbf, nX);
  init_state<<<(B_ * KC + 255) / 256, 256, 0, stream>>>(xbf, Abuf, c, hf);

  for (int t = 0; t < T_; ++t) {
    lstm_step_gemm<<<dim3(64, 2), 256, 0, stream>>>(Abuf, Bp, z0, z1);
    lstm_elem<<<(B_ * H_ + 255) / 256, 256, 0, stream>>>(z0, z1, bias, c, Abuf, hf, xbf, t + 1);
  }

  final_dense<<<B_, 256, 0, stream>>>(hf, dw, db, out);
}
